// CyclicPositional_83313775608157
// MI455X (gfx1250) — matrix operand order repaired and checked by reading, not compiled, not run on hardware
//
#include <hip/hip_runtime.h>

typedef __attribute__((ext_vector_type(16))) _Float16 v16h;
typedef __attribute__((ext_vector_type(8)))  float    v8f;

#define B_   16
#define D_   1024
#define K_   64
#define TWOK 128
#define NMAX 8192
#define LDS_STRIDE 65   // 64 + 1 pad -> conflict-free gather across rows

union AFrag { v16h h; unsigned u[8]; };
union BFrag { v16h h; uint4 q[2]; };
union CFrag { v8f  f; float x[8]; };

// Pre-pass: convert W (f32 [1024,128]) -> f16 copy in scratch, and emit the
// [B,N] mask (as f32 0/1) into the tail of d_out. 131072 threads do both.
__global__ __launch_bounds__(256)
void CyclicPositional_prep_kernel(const float* __restrict__ W,
                                  const int* __restrict__ lengths,
                                  _Float16* __restrict__ Wh,
                                  float* __restrict__ maskOut) {
    int i = blockIdx.x * 256 + threadIdx.x;   // 0 .. 131071 (== D*2K == B*N)
    Wh[i] = (_Float16)W[i];
    int b = i >> 13;            // / NMAX
    int n = i & (NMAX - 1);
    __builtin_nontemporal_store((n < lengths[b]) ? 1.0f : 0.0f, &maskOut[i]);
}

// Main kernel: one block per (b, 16-position tile). 8 waves cover all D=1024.
__global__ __launch_bounds__(256)
void CyclicPositional_posemb_kernel(const int* __restrict__ lengths,
                                    const _Float16* __restrict__ Wh,
                                    float* __restrict__ out) {
    __shared__ unsigned cs[16 * LDS_STRIDE];  // packed (cos,sin) f16x2 per (m,k)

    const int tid   = threadIdx.x;
    const int ntile = blockIdx.x;             // 0..511
    const int b     = blockIdx.y;             // 0..15
    const int n0    = ntile * 16;
    const int len   = lengths[b];
    const float lenf = (float)len;

    // ---- Fill feature tile (16 rows x 64 freqs), 4 entries per thread ----
    {
        int base = tid * 4;                   // same m, 4 consecutive k
        int m  = base >> 6;
        int k0 = base & 63;
        int n  = n0 + m;
        float nf  = (float)n;
        float amp = (n < len) ? 0.125f : 0.0f;   // 1/sqrt(K), masked
        #pragma unroll
        for (int i = 0; i < 4; ++i) {
            float kk   = (float)(k0 + i + 1);
            float prod = nf * kk;                       // exact: < 2^24
            float q    = floorf(prod / lenf);
            float r    = prod - q * lenf;               // prod mod len, ~exact
            float theta = 6.28318530717958647692f * (r / lenf);
            float sn, csn;
            __sincosf(theta, &sn, &csn);
            union { _Float16 h[2]; unsigned u; } pk;
            pk.h[0] = (_Float16)(csn * amp);            // even f = cos (low half)
            pk.h[1] = (_Float16)(sn  * amp);            // odd  f = sin (high half)
            cs[m * LDS_STRIDE + k0 + i] = pk.u;
        }
    }
    __syncthreads();

    const int lane = tid & 31;
    const int wave = tid >> 5;
    const int hi   = lane >> 4;               // half-wave select
    const int lm   = lane & 15;

    // ---- Build 4 A fragments (16x32 f16 each, K-chunks of 32 halves) ----
    // Layout: lane<16 VGPR j -> packed k=j ; lanes>=16 add +8 (contiguous, matches B).
    AFrag a[4];
    #pragma unroll
    for (int c = 0; c < 4; ++c) {
        #pragma unroll
        for (int j = 0; j < 8; ++j) {
            int kidx = 16 * c + j + hi * 8;
            a[c].u[j] = cs[lm * LDS_STRIDE + kidx];
        }
    }

    float* outBase = out + ((size_t)(b * NMAX + n0)) * D_;

    // ---- 8 sub-tiles of 16 columns; wave covers d in [wave*128, wave*128+128)
    #pragma unroll 1
    for (int s = 0; s < 8; ++s) {
        int d = wave * 128 + s * 16;
        v8f acc = {};
        #pragma unroll
        for (int c = 0; c < 4; ++c) {
            // B (32x16): lane = column (lm), K range selected by half-wave.
            // W row-major [d, f] -> 16 contiguous halves = two b128 loads.
            BFrag bb;
            const uint4* p = (const uint4*)(Wh + (size_t)(d + lm) * TWOK
                                               + c * 32 + hi * 16);
            bb.q[0] = p[0];
            bb.q[1] = p[1];
            acc = __builtin_amdgcn_wmma_f32_16x16x32_f16(
                      /*neg_a=*/false, a[c].h, /*neg_b=*/false, bb.h,
                      /*c_mod=*/(short)0, acc, /*reuse_a=*/false, /*reuse_b=*/false);
        }
        // C layout: lane<16 VGPR j -> row j; lanes>=16 -> row j+8; col = lm.
        CFrag cf; cf.f = acc;
        #pragma unroll
        for (int j = 0; j < 8; ++j) {
            int row = hi * 8 + j;
            __builtin_nontemporal_store(cf.x[j],
                &outBase[(size_t)row * D_ + d + lm]);
        }
    }
}

extern "C" void kernel_launch(void* const* d_in, const int* in_sizes, int n_in,
                              void* d_out, int out_size, void* d_ws, size_t ws_size,
                              hipStream_t stream) {
    const int*   lengths = (const int*)d_in[0];
    const float* W       = (const float*)d_in[1];
    float* out     = (float*)d_out;
    float* maskOut = out + (size_t)B_ * NMAX * D_;   // tuple output #2
    _Float16* Wh   = (_Float16*)d_ws;                // 256 KB f16 copy of W

    // Pre-pass: W f32->f16 + mask (131072 threads = D*2K = B*N)
    CyclicPositional_prep_kernel<<<dim3(512), dim3(256), 0, stream>>>(
        W, lengths, Wh, maskOut);

    // Main GEMM: grid = (N/16 tiles, B), 256 threads = 8 wave32
    CyclicPositional_posemb_kernel<<<dim3(NMAX / 16, B_), dim3(256), 0, stream>>>(
        lengths, Wh, out);
}
